// LSTM_EDC_Model_6717328851494
// MI455X (gfx1250) — compile-verified, matrix operand order checked by reading, not run on hardware
//
#include <hip/hip_runtime.h>
#include <hip/hip_bf16.h>

// ---------------------------------------------------------------------------
// Fused 2-layer LSTM (H=128, B=2 padded to 16 rows, T=144000) for gfx1250.
// Single workgroup, 8 waves (wave32). Wave w owns hidden units [16w,16w+16).
// Layer math done with v_wmma_f32_16x16x32_f16 (f16 operands, f32 accum).
//  - layer0 recurrent weights: packed f16 B-fragments held in VGPRs
//  - layer1 [w_ih1|w_hh1]    : packed f16 B-fragments staged in LDS (256KB)
//  - h0/h1 broadcast per step through a double-buffered LDS tile
//    (16 rows x 264 f16, padded stride to spread LDS banks)
//  - xproj0 (time-invariant input projection of layer0) precomputed and used
//    as the WMMA C initializer; biases of layer1 likewise.
// Fragment layouts follow CDNA5 ISA 16-bit A/B tables:
//   lane<16 : rows/cols 0..15, k = {0..7} then {16..23}
//   lane>=16: same rows/cols,   k = {8..15} then {24..31}
// ---------------------------------------------------------------------------

typedef __attribute__((ext_vector_type(16))) _Float16 v16h;
typedef __attribute__((ext_vector_type(8)))  float    v8f;

#define HID       128
#define GATES     512
#define TSTEPS    144000
#define HROW      264          // padded f16 row stride for the h LDS tile
#define NBANDS    10

// d_ws layout (bytes)
#define WS_FRAG0  0            // layer0:  128 frags * 512 f16 = 131072 B
#define WS_FRAG1  131072       // layer1:  256 frags * 512 f16 = 262144 B
#define WS_FRAGO  393216       // head:      4 frags * 512 f16 =   4096 B
#define WS_XPROJ  397312       // xproj0:  16*512 f32          =  32768 B
// total = 430080 B

__device__ __forceinline__ v8f wmma32(v16h a, v16h b, v8f c) {
  return __builtin_amdgcn_wmma_f32_16x16x32_f16(false, a, false, b,
                                                (short)0, c, false, false);
}

// contiguous 32B-per-lane fragment load (packed B fragments)
__device__ __forceinline__ v16h ld_frag16(const _Float16* p) {
  union { uint4 u[2]; v16h h; } u;
  u.u[0] = *(const uint4*)p;
  u.u[1] = *(const uint4*)(p + 8);
  return u.h;
}

// A-fragment from row-major f16 LDS tile (row stride HROW), tile K0..K0+31
__device__ __forceinline__ v16h ld_afrag(const _Float16* row0, int K0, int lane) {
  const int row  = lane & 15;
  const int half = lane >> 4;
  const _Float16* p = row0 + row * HROW + K0 + 8 * half;
  union { uint4 u[2]; v16h h; } u;
  u.u[0] = *(const uint4*)p;          // k = K0 + 8*half + {0..7}
  u.u[1] = *(const uint4*)(p + 16);   // k = K0 + 8*half + {16..23}
  return u.h;
}

// store a 16x16 f32 D fragment (this wave's 16 hidden cols) as f16 rows
__device__ __forceinline__ void st_hfrag(_Float16* row0, int colbase, int lane,
                                         const float* h8) {
  const int n     = lane & 15;
  const int rbase = (lane >> 4) * 8;
#pragma unroll
  for (int r = 0; r < 8; ++r)
    row0[(rbase + r) * HROW + colbase + n] = (_Float16)h8[r];
}

__device__ __forceinline__ float sig_fast(float x) {
  x = fminf(fmaxf(x, -30.f), 30.f);
  float e = __expf(-x);
  return __builtin_amdgcn_rcpf(1.f + e);
}
__device__ __forceinline__ float tanh_fast(float x) {
  x = fminf(fmaxf(x, -15.f), 15.f);
  float e = __expf(-2.f * x);
  return (1.f - e) * __builtin_amdgcn_rcpf(1.f + e);
}

__device__ __forceinline__ void lstm_cell(const v8f& gi, const v8f& gf,
                                          const v8f& gg, const v8f& go,
                                          v8f& c, float* h8) {
#pragma unroll
  for (int j = 0; j < 8; ++j) {
    float i  = sig_fast(gi[j]);
    float f  = sig_fast(gf[j]);
    float g  = tanh_fast(gg[j]);
    float o  = sig_fast(go[j]);
    float cn = f * c[j] + i * g;
    c[j]  = cn;
    h8[j] = o * tanh_fast(cn);
  }
}

// ------------------------- prep kernel: encoder MLP + xproj0 ----------------
__global__ void prep_encoder(const float* __restrict__ x,
                             const float* __restrict__ w1, const float* __restrict__ b1,
                             const float* __restrict__ w2, const float* __restrict__ b2,
                             const float* __restrict__ w_ih0,
                             const float* __restrict__ b_ih0,
                             const float* __restrict__ b_hh0,
                             float* __restrict__ xproj0 /*16x512*/) {
  __shared__ float hid[2][64];
  __shared__ float lat[2][128];
  int tid = threadIdx.x;
  if (tid < 128) {
    int b = tid >> 6, j = tid & 63;
    float s = b1[j];
    for (int k = 0; k < 16; ++k) s += x[b * 16 + k] * w1[j * 16 + k];
    hid[b][j] = fmaxf(s, 0.f);
  }
  __syncthreads();
  {
    int b = tid >> 7, j = tid & 127;
    float s = b2[j];
    for (int k = 0; k < 64; ++k) s += hid[b][k] * w2[j * 64 + k];
    lat[b][j] = fmaxf(s, 0.f);
  }
  __syncthreads();
  for (int n = tid; n < GATES; n += 256) {
    float base = b_ih0[n] + b_hh0[n];
    for (int row = 0; row < 16; ++row) {
      float s = base;
      if (row < 2)
        for (int k = 0; k < HID; ++k) s += lat[row][k] * w_ih0[n * HID + k];
      xproj0[row * GATES + n] = s;   // rows>=2: phantom batch, finite
    }
  }
}

// ----------------- pack kernel: fp32 weights -> f16 WMMA B fragments --------
__global__ void pack_weights(const float* __restrict__ w_hh0,
                             const float* __restrict__ w_ih1,
                             const float* __restrict__ w_hh1,
                             const float* __restrict__ w_out,
                             _Float16* __restrict__ frag0,
                             _Float16* __restrict__ frag1,
                             _Float16* __restrict__ fragO) {
  int gid  = blockIdx.x * blockDim.x + threadIdx.x;  // one thread per (frag,lane)
  int lane = gid & 31;
  int f    = gid >> 5;
  if (f >= 128 + 256 + 4) return;
  int half = (lane >> 4) & 1;
  int nl   = lane & 15;
  if (f < 128) {                       // layer0: gate-tile gt, k-tile kt
    int gt = f >> 2, kt = f & 3;
    int n  = gt * 16 + nl;
    _Float16* dst = frag0 + ((size_t)f * 32 + lane) * 16;
#pragma unroll
    for (int e = 0; e < 16; ++e) {
      int k = kt * 32 + 8 * half + e + (e >= 8 ? 8 : 0);
      dst[e] = (_Float16)w_hh0[n * HID + k];
    }
  } else if (f < 128 + 256) {          // layer1: K = [x(128) | h(128)]
    int f1 = f - 128;
    int gt = f1 >> 3, kt = f1 & 7;
    int n  = gt * 16 + nl;
    _Float16* dst = frag1 + ((size_t)f1 * 32 + lane) * 16;
#pragma unroll
    for (int e = 0; e < 16; ++e) {
      int k = kt * 32 + 8 * half + e + (e >= 8 ? 8 : 0);
      float v = (k < HID) ? w_ih1[n * HID + k] : w_hh1[n * HID + (k - HID)];
      dst[e] = (_Float16)v;
    }
  } else {                             // output head (10 of 16 cols used)
    int kt = f - 384;
    int n  = nl;
    _Float16* dst = fragO + ((size_t)kt * 32 + lane) * 16;
#pragma unroll
    for (int e = 0; e < 16; ++e) {
      int k = kt * 32 + 8 * half + e + (e >= 8 ? 8 : 0);
      float v = (n < NBANDS) ? w_out[n * HID + k] : 0.f;
      dst[e] = (_Float16)v;
    }
  }
}

// ----------------------------- main recurrent kernel ------------------------
__global__ void __launch_bounds__(256, 1)
lstm_main(const _Float16* __restrict__ frag0,
          const _Float16* __restrict__ frag1,
          const _Float16* __restrict__ fragO,
          const float* __restrict__ xproj0,
          const float* __restrict__ b_ih1,
          const float* __restrict__ b_hh1,
          const float* __restrict__ b_out,
          float* __restrict__ out) {
  extern __shared__ __align__(16) _Float16 smem[];
  _Float16* sW1 = smem;                       // 256 frags * 512 f16 = 256 KB
  _Float16* hc  = smem + 256 * 512;           // 2 buffers * 16*HROW f16

  const int tid  = threadIdx.x;
  const int lane = tid & 31;
  const int wv   = tid >> 5;

  // stage layer1 fragments into LDS (256 KB, once)
  {
    const uint4* src = (const uint4*)frag1;
    uint4*       dst = (uint4*)sW1;
    for (int i = tid; i < (256 * 512 * 2) / 16; i += 256) dst[i] = src[i];
  }
  // zero both h buffers (h0(-1)=h1(-1)=0, plus padding columns)
  for (int i = tid; i < 2 * 16 * HROW; i += 256) hc[i] = (_Float16)0.f;

  // layer0 recurrent-weight fragments -> registers (16 frags/wave)
  v16h b0[4][4];
#pragma unroll
  for (int g = 0; g < 4; ++g)
#pragma unroll
    for (int kt = 0; kt < 4; ++kt) {
      int f = ((g * 8 + wv) * 4 + kt);
      b0[g][kt] = ld_frag16(frag0 + ((size_t)f * 32 + lane) * 16);
    }

  // output-head fragments (used by wave 0 only)
  v16h wo[4];
#pragma unroll
  for (int kt = 0; kt < 4; ++kt)
    wo[kt] = ld_frag16(fragO + ((size_t)kt * 32 + lane) * 16);
  float bo = (lane < NBANDS) ? b_out[lane] : 0.f;

  // xproj0 fragments (C init of layer0) and layer1 bias fragments
  v8f xp[4], bs1[4];
#pragma unroll
  for (int g = 0; g < 4; ++g) {
    int n   = (g * 8 + wv) * 16 + (lane & 15);
    float b = b_ih1[n] + b_hh1[n];
#pragma unroll
    for (int r = 0; r < 8; ++r) {
      xp[g][r]  = xproj0[((lane >> 4) * 8 + r) * GATES + n];
      bs1[g][r] = b;
    }
  }

  v8f c0 = {0, 0, 0, 0, 0, 0, 0, 0};
  v8f c1 = {0, 0, 0, 0, 0, 0, 0, 0};
  const int cb = 16 * wv;   // this wave's hidden-column base
  __syncthreads();

#pragma clang loop unroll(disable)
  for (int t = 0; t < TSTEPS; ++t) {
    _Float16* wbuf = hc + (t & 1) * (16 * HROW);
    _Float16* rbuf = hc + ((t & 1) ^ 1) * (16 * HROW);

    // ---- layer 0: gates = xproj0 + h0(t-1) @ w_hh0^T ----
    v8f ai = xp[0], af = xp[1], ag = xp[2], ao = xp[3];
#pragma unroll
    for (int kt = 0; kt < 4; ++kt) {
      v16h a = ld_afrag(rbuf, 32 * kt, lane);
      ai = wmma32(a, b0[0][kt], ai);
      af = wmma32(a, b0[1][kt], af);
      ag = wmma32(a, b0[2][kt], ag);
      ao = wmma32(a, b0[3][kt], ao);
    }
    float h8[8];
    lstm_cell(ai, af, ag, ao, c0, h8);
    st_hfrag(wbuf, cb, lane, h8);       // h0(t) -> wbuf cols [0,128)
    __syncthreads();

    // ---- layer 1: gates = bias + [h0(t)|h1(t-1)] @ [w_ih1|w_hh1]^T ----
    ai = bs1[0]; af = bs1[1]; ag = bs1[2]; ao = bs1[3];
#pragma unroll
    for (int kt = 0; kt < 8; ++kt) {
      v16h a = ld_afrag((kt < 4) ? wbuf : rbuf, 32 * kt, lane);
      ai = wmma32(a, ld_frag16(sW1 + ((size_t)((0 * 8 + wv) * 8 + kt) * 32 + lane) * 16), ai);
      af = wmma32(a, ld_frag16(sW1 + ((size_t)((1 * 8 + wv) * 8 + kt) * 32 + lane) * 16), af);
      ag = wmma32(a, ld_frag16(sW1 + ((size_t)((2 * 8 + wv) * 8 + kt) * 32 + lane) * 16), ag);
      ao = wmma32(a, ld_frag16(sW1 + ((size_t)((3 * 8 + wv) * 8 + kt) * 32 + lane) * 16), ao);
    }
    lstm_cell(ai, af, ag, ao, c1, h8);
    st_hfrag(wbuf, HID + cb, lane, h8); // h1(t) -> wbuf cols [128,256)
    __syncthreads();

    // ---- output head (wave 0): out[t] = h1(t) @ w_out^T + b_out ----
    if (wv == 0) {
      v8f acc = {0, 0, 0, 0, 0, 0, 0, 0};
#pragma unroll
      for (int kt = 0; kt < 4; ++kt) {
        v16h a = ld_afrag(wbuf, HID + 32 * kt, lane);
        acc = wmma32(a, wo[kt], acc);
      }
      if (lane < NBANDS) {              // rows 0,1 = batch 0,1 (elements 0,1)
        out[(size_t)t * NBANDS + lane]            = acc[0] + bo;
        out[((size_t)TSTEPS + t) * NBANDS + lane] = acc[1] + bo;
      }
    }
  }
}

// ---------------------------------------------------------------------------
extern "C" void kernel_launch(void* const* d_in, const int* in_sizes, int n_in,
                              void* d_out, int out_size, void* d_ws, size_t ws_size,
                              hipStream_t stream) {
  const float* x     = (const float*)d_in[0];
  const float* w1    = (const float*)d_in[1];
  const float* b1    = (const float*)d_in[2];
  const float* w2    = (const float*)d_in[3];
  const float* b2    = (const float*)d_in[4];
  const float* w_ih0 = (const float*)d_in[5];
  const float* w_hh0 = (const float*)d_in[6];
  const float* b_ih0 = (const float*)d_in[7];
  const float* b_hh0 = (const float*)d_in[8];
  const float* w_ih1 = (const float*)d_in[9];
  const float* w_hh1 = (const float*)d_in[10];
  const float* b_ih1 = (const float*)d_in[11];
  const float* b_hh1 = (const float*)d_in[12];
  const float* w_out = (const float*)d_in[13];
  const float* b_out = (const float*)d_in[14];

  char*      ws     = (char*)d_ws;
  _Float16*  frag0  = (_Float16*)(ws + WS_FRAG0);
  _Float16*  frag1  = (_Float16*)(ws + WS_FRAG1);
  _Float16*  fragO  = (_Float16*)(ws + WS_FRAGO);
  float*     xproj0 = (float*)(ws + WS_XPROJ);

  prep_encoder<<<1, 256, 0, stream>>>(x, w1, b1, w2, b2, w_ih0, b_ih0, b_hh0, xproj0);
  pack_weights<<<(388 * 32 + 255) / 256, 256, 0, stream>>>(w_hh0, w_ih1, w_hh1,
                                                           w_out, frag0, frag1, fragO);

  const size_t smem = (size_t)256 * 512 * 2 + (size_t)2 * 16 * HROW * 2;  // 279040 B
  (void)hipFuncSetAttribute((const void*)lstm_main,
                            hipFuncAttributeMaxDynamicSharedMemorySize, (int)smem);
  lstm_main<<<1, 256, smem, stream>>>(frag0, frag1, fragO, xproj0,
                                      b_ih1, b_hh1, b_out, (float*)d_out);
  (void)in_sizes; (void)n_in; (void)out_size; (void)ws_size;
}
